// FlickrSAGE_59717225283873
// MI455X (gfx1250) — matrix operand order repaired and checked by reading, not compiled, not run on hardware
//
#include <hip/hip_runtime.h>

#define N_NODES 89250

// ---- WMMA fragment types (wave32, gfx1250) -------------------------------
typedef _Float16 v16h __attribute__((ext_vector_type(16)));
typedef float    v8f  __attribute__((ext_vector_type(8)));
typedef unsigned int v4u __attribute__((ext_vector_type(4)));

union HFrag { v16h h; v4u u[2]; };

// A-matrix 16x32 f16 layout: lanes 0-15 row M=lane, lanes 16-31 row M=lane-16.
// Lane half 0: vector elems 0..7 = K 0..7, elems 8..15 = K 16..23.
// Lane half 1: vector elems 0..7 = K 8..15, elems 8..15 = K 24..31.
__device__ __forceinline__ v16h load_a_frag(const _Float16* rowp, int kb, int khalf) {
  HFrag f;
  f.u[0] = *(const v4u*)(rowp + kb + khalf * 8);
  f.u[1] = *(const v4u*)(rowp + kb + 16 + khalf * 8);
  return f.h;
}

// B-matrix 32x16 f16 layout (same K order as A): lane half 0 elems 0..7 = K 0..7,
// elems 8..15 = K 16..23; lane half 1 elems 0..7 = K 8..15, elems 8..15 = K 24..31.
// rowp points at W[col] row (row-major [M,Kpad]) so B[k][col] = rowp[k].
__device__ __forceinline__ v16h load_b_frag(const _Float16* rowp, int kb, int khalf) {
  HFrag f;
  f.u[0] = *(const v4u*)(rowp + kb + khalf * 8);
  f.u[1] = *(const v4u*)(rowp + kb + 16 + khalf * 8);
  return f.h;
}

__device__ __forceinline__ v8f wmma16(v16h a, v16h b, v8f c) {
  return __builtin_amdgcn_wmma_f32_16x16x32_f16(false, a, false, b, (short)0,
                                                c, false, false);
}

// ---- graph degree (computed once, reused for all 3 layers) ----------------
__global__ void degree_kernel(const int* __restrict__ dst, float* __restrict__ deg, int E) {
  int e = blockIdx.x * 256 + threadIdx.x;
  if (e < E) {
#if defined(__HIP_PLATFORM_AMD__)
    unsafeAtomicAdd(&deg[dst[e]], 1.0f);
#else
    atomicAdd(&deg[dst[e]], 1.0f);
#endif
  }
}

// ---- per-edge gather + scatter-add: agg[dst] += x[src] --------------------
// One block per edge; float4 gather of the source row, hardware f32 atomics
// into agg (agg <= 183 MB, resident in the 192 MB L2 -> on-chip RMW).
__global__ __launch_bounds__(128) void scatter_kernel(
    const float* __restrict__ x, const int* __restrict__ src,
    const int* __restrict__ dst, float* __restrict__ agg, int C4, int E) {
  int e = blockIdx.x;
  if (e >= E) return;
  int s = src[e];
  int d = dst[e];
  const float4* xs = (const float4*)x + (size_t)s * C4;
  float* ag = agg + (size_t)d * ((size_t)C4 * 4);
  for (int t = threadIdx.x; t < C4; t += 128) {
    float4 v = xs[t];
#if defined(__HIP_PLATFORM_AMD__)
    unsafeAtomicAdd(ag + 4 * t + 0, v.x);
    unsafeAtomicAdd(ag + 4 * t + 1, v.y);
    unsafeAtomicAdd(ag + 4 * t + 2, v.z);
    unsafeAtomicAdd(ag + 4 * t + 3, v.w);
#else
    atomicAdd(ag + 4 * t + 0, v.x);
    atomicAdd(ag + 4 * t + 1, v.y);
    atomicAdd(ag + 4 * t + 2, v.z);
    atomicAdd(ag + 4 * t + 3, v.w);
#endif
  }
}

// ---- normalize (mean = agg / clamp(deg,1)) and f32->f16 with K padding ----
__global__ void convert_mean_kernel(const float* __restrict__ agg,
                                    const float* __restrict__ deg,
                                    _Float16* __restrict__ A, int C, int Kpad) {
  size_t idx = (size_t)blockIdx.x * 256 + threadIdx.x;
  size_t total = (size_t)N_NODES * Kpad;
  if (idx >= total) return;
  int k = (int)(idx % Kpad);
  size_t n = idx / Kpad;
  float v = 0.0f;
  if (k < C) {
    float d = deg[n];
    d = d < 1.0f ? 1.0f : d;
    v = agg[n * C + k] / d;
  }
  A[idx] = (_Float16)v;
}

__global__ void convert_feat_kernel(const float* __restrict__ xin,
                                    _Float16* __restrict__ A, int C, int Kpad) {
  size_t idx = (size_t)blockIdx.x * 256 + threadIdx.x;
  size_t total = (size_t)N_NODES * Kpad;
  if (idx >= total) return;
  int k = (int)(idx % Kpad);
  size_t n = idx / Kpad;
  float v = (k < C) ? xin[n * C + k] : 0.0f;
  A[idx] = (_Float16)v;
}

// ---- weight f32->f16 with K padding and M padding (for the 7-wide layer) --
__global__ void convert_w_kernel(const float* __restrict__ w,
                                 _Float16* __restrict__ W, int Cout, int Cin,
                                 int Kpad, int Mpad) {
  size_t idx = (size_t)blockIdx.x * 256 + threadIdx.x;
  size_t total = (size_t)Mpad * Kpad;
  if (idx >= total) return;
  int k = (int)(idx % Kpad);
  int m = (int)(idx / Kpad);
  float v = (m < Cout && k < Cin) ? w[(size_t)m * Cin + k] : 0.0f;
  W[idx] = (_Float16)v;
}

// ---- fused SAGE GEMM: out = Al*Wl^T + Ar*Wr^T + b (+ReLU) via WMMA --------
// Each wave owns a 32(node) x 32(feature) tile = 2x2 accumulators, so every
// A/B fragment feeds two WMMAs (halves load traffic per FLOP) and the four
// independent accumulator chains hide WMMA->WMMA latency. 8 waves per block.
__global__ __launch_bounds__(256) void sage_wmma_gemm(
    const _Float16* __restrict__ Al, const _Float16* __restrict__ Ar,
    const _Float16* __restrict__ Wl, const _Float16* __restrict__ Wr,
    const float* __restrict__ bias, float* __restrict__ out,
    int Nn, int Kpad, int Cout, int relu) {
  const int lane = threadIdx.x & 31;
  const int wave = threadIdx.x >> 5;
  const int nodeBase = (blockIdx.x * 8 + wave) * 32;
  if (nodeBase >= Nn) return;  // wave-uniform: EXEC stays all-ones for WMMA
  const int mBase = blockIdx.y * 32;

  const int row = lane & 15;     // A row within subtile / B column
  const int khalf = lane >> 4;   // which K half this lane supplies

  int r0 = nodeBase + row;       // clamp loads; stores are guarded below
  if (r0 >= Nn) r0 = Nn - 1;
  int r1 = nodeBase + 16 + row;
  if (r1 >= Nn) r1 = Nn - 1;

  const _Float16* aL0 = Al + (size_t)r0 * Kpad;
  const _Float16* aL1 = Al + (size_t)r1 * Kpad;
  const _Float16* aR0 = Ar + (size_t)r0 * Kpad;
  const _Float16* aR1 = Ar + (size_t)r1 * Kpad;
  const _Float16* wL0 = Wl + (size_t)(mBase + row) * Kpad;
  const _Float16* wL1 = Wl + (size_t)(mBase + 16 + row) * Kpad;
  const _Float16* wR0 = Wr + (size_t)(mBase + row) * Kpad;
  const _Float16* wR1 = Wr + (size_t)(mBase + 16 + row) * Kpad;

  v8f acc00 = {}, acc01 = {}, acc10 = {}, acc11 = {};
  for (int kb = 0; kb < Kpad; kb += 32) {
    // left chain: mean @ w_l^T
    v16h a0 = load_a_frag(aL0, kb, khalf);
    v16h a1 = load_a_frag(aL1, kb, khalf);
    v16h b0 = load_b_frag(wL0, kb, khalf);
    v16h b1 = load_b_frag(wL1, kb, khalf);
    acc00 = wmma16(a0, b0, acc00);
    acc01 = wmma16(a0, b1, acc01);
    acc10 = wmma16(a1, b0, acc10);
    acc11 = wmma16(a1, b1, acc11);
    // right chain: x @ w_r^T
    a0 = load_a_frag(aR0, kb, khalf);
    a1 = load_a_frag(aR1, kb, khalf);
    b0 = load_b_frag(wR0, kb, khalf);
    b1 = load_b_frag(wR1, kb, khalf);
    acc00 = wmma16(a0, b0, acc00);
    acc01 = wmma16(a0, b1, acc01);
    acc10 = wmma16(a1, b0, acc10);
    acc11 = wmma16(a1, b1, acc11);
  }

  const int m0 = mBase + row;
  const int m1 = mBase + 16 + row;
  float bv0 = (m0 < Cout) ? bias[m0] : 0.0f;
  float bv1 = (m1 < Cout) ? bias[m1] : 0.0f;
  #pragma unroll
  for (int i = 0; i < 8; ++i) {
    // C/D layout: VGPR i -> M=i for lanes 0-15, M=8+i for lanes 16-31
    int n0 = nodeBase + khalf * 8 + i;
    int n1 = n0 + 16;
    if (n0 < Nn) {
      if (m0 < Cout) {
        float v = acc00[i] + bv0;
        if (relu) v = fmaxf(v, 0.0f);
        out[(size_t)n0 * Cout + m0] = v;
      }
      if (m1 < Cout) {
        float v = acc01[i] + bv1;
        if (relu) v = fmaxf(v, 0.0f);
        out[(size_t)n0 * Cout + m1] = v;
      }
    }
    if (n1 < Nn) {
      if (m0 < Cout) {
        float v = acc10[i] + bv0;
        if (relu) v = fmaxf(v, 0.0f);
        out[(size_t)n1 * Cout + m0] = v;
      }
      if (m1 < Cout) {
        float v = acc11[i] + bv1;
        if (relu) v = fmaxf(v, 0.0f);
        out[(size_t)n1 * Cout + m1] = v;
      }
    }
  }
}

// ---- one SAGE layer -------------------------------------------------------
static void run_layer(const float* xin, int Cin, int Kpad, int Cout, int Mpad,
                      const float* wl, const float* wr, const float* bias,
                      const int* src, const int* dst, int E, float* agg,
                      const float* deg, _Float16* Al, _Float16* Ar,
                      _Float16* Wlb, _Float16* Wrb, float* out, int relu,
                      hipStream_t stream) {
  const int N = N_NODES;
  hipMemsetAsync(agg, 0, (size_t)N * Cin * sizeof(float), stream);
  scatter_kernel<<<E, 128, 0, stream>>>(xin, src, dst, agg, Cin / 4, E);

  size_t totalA = (size_t)N * Kpad;
  int gA = (int)((totalA + 255) / 256);
  convert_mean_kernel<<<gA, 256, 0, stream>>>(agg, deg, Al, Cin, Kpad);
  convert_feat_kernel<<<gA, 256, 0, stream>>>(xin, Ar, Cin, Kpad);

  size_t totalW = (size_t)Mpad * Kpad;
  int gW = (int)((totalW + 255) / 256);
  convert_w_kernel<<<gW, 256, 0, stream>>>(wl, Wlb, Cout, Cin, Kpad, Mpad);
  convert_w_kernel<<<gW, 256, 0, stream>>>(wr, Wrb, Cout, Cin, Kpad, Mpad);

  int nodeTiles = (N + 31) / 32;              // 32-node wave tiles
  dim3 grid((nodeTiles + 7) / 8, Mpad / 32);  // 8 waves per block
  sage_wmma_gemm<<<grid, 256, 0, stream>>>(Al, Ar, Wlb, Wrb, bias, out, N,
                                           Kpad, Cout, relu);
}

extern "C" void kernel_launch(void* const* d_in, const int* in_sizes, int n_in,
                              void* d_out, int out_size, void* d_ws, size_t ws_size,
                              hipStream_t stream) {
  (void)n_in; (void)out_size; (void)ws_size;
  const float* x    = (const float*)d_in[0];
  const int*   ei   = (const int*)d_in[1];
  const int    E    = in_sizes[1] / 2;
  const int*   src  = ei;
  const int*   dst  = ei + E;
  const float* w_l1 = (const float*)d_in[2];
  const float* w_r1 = (const float*)d_in[3];
  const float* b1   = (const float*)d_in[4];
  const float* w_l2 = (const float*)d_in[5];
  const float* w_r2 = (const float*)d_in[6];
  const float* b2   = (const float*)d_in[7];
  const float* w_l3 = (const float*)d_in[8];
  const float* w_r3 = (const float*)d_in[9];
  const float* b3   = (const float*)d_in[10];

  const int N = N_NODES;
  char* p = (char*)d_ws;
  auto carve = [&](size_t bytes) {
    char* r = p;
    p += (bytes + 511) & ~(size_t)511;
    return r;
  };
  float*    deg  = (float*)carve((size_t)N * 4);            // node degrees
  float*    agg  = (float*)carve((size_t)N * 512 * 4);      // f32 scatter target
  _Float16* Al   = (_Float16*)carve((size_t)N * 512 * 2);   // mean, f16, K-padded
  _Float16* Ar   = (_Float16*)carve((size_t)N * 512 * 2);   // self feats, f16
  _Float16* Wlb  = (_Float16*)carve((size_t)512 * 512 * 2); // w_l f16 padded
  _Float16* Wrb  = (_Float16*)carve((size_t)512 * 512 * 2); // w_r f16 padded
  float*    hbuf = (float*)carve((size_t)N * 512 * 4);      // hidden (reused)

  // degree once (graph topology is shared by all 3 layers)
  hipMemsetAsync(deg, 0, (size_t)N * 4, stream);
  degree_kernel<<<(E + 255) / 256, 256, 0, stream>>>(dst, deg, E);

  // layer 1: 500 -> 512, ReLU
  run_layer(x, 500, 512, 512, 512, w_l1, w_r1, b1, src, dst, E, agg, deg, Al,
            Ar, Wlb, Wrb, hbuf, 1, stream);
  // layer 2: 512 -> 256, ReLU (output aliases hbuf; GEMM only reads f16 copies)
  run_layer(hbuf, 512, 512, 256, 256, w_l2, w_r2, b2, src, dst, E, agg, deg,
            Al, Ar, Wlb, Wrb, hbuf, 1, stream);
  // layer 3: 256 -> 7 (M padded to 32 for uniform tiling), no ReLU -> d_out
  run_layer(hbuf, 256, 256, 7, 32, w_l3, w_r3, b3, src, dst, E, agg, deg, Al,
            Ar, Wlb, Wrb, (float*)d_out, 0, stream);
}
